// PointerGenerator_11639361372585
// MI455X (gfx1250) — matrix operand order repaired and checked by reading, not compiled, not run on hardware
//
#include <hip/hip_runtime.h>
#include <math.h>

// ---------------------------------------------------------------------------
// Pointer-generator network for MI455X (gfx1250), wave32 + WMMA bf16.
// All matrix math uses v_wmma_f32_16x16x32_bf16. Heavy GEMM (vocab projection,
// 16x50000x256 per decoder step) stays L2-resident (l2W bf16 = 25.6MB << 192MB L2).
// ---------------------------------------------------------------------------

typedef __attribute__((ext_vector_type(16))) __bf16 v16bf;
typedef __attribute__((ext_vector_type(8)))  float  v8f;
typedef __attribute__((ext_vector_type(8)))  unsigned short ushort8;
typedef unsigned short u16;

union FragU { u16 u[16]; ushort8 u8[2]; v16bf bf; };

__device__ __forceinline__ u16 f2bf(float f){
  union { float f; unsigned u; } v; v.f = f;
  unsigned r = v.u + 0x7FFFu + ((v.u >> 16) & 1u);
  return (u16)(r >> 16);
}
__device__ __forceinline__ float sigm(float x){ return 1.f / (1.f + expf(-x)); }

// A fragment (16x32 bf16, MxK). ISA 7.12.2: lanes 0-15: m=lane, K chunks
// [k0..k0+7] and [k0+16..k0+23]; lanes 16-31: m=lane-16, chunks +8.
__device__ __forceinline__ v16bf frag_a(const u16* A, int lda, int m0, int k0){
  int lane = threadIdx.x & 31;
  const u16* p = A + (size_t)(m0 + (lane & 15)) * lda + k0 + ((lane >> 4) << 3);
  FragU f;
  f.u8[0] = *(const ushort8*)p;
  f.u8[1] = *(const ushort8*)(p + 16);
  return f.bf;
}
// B fragment (32x16 bf16, KxN) built from W stored row-major (N x K): C = A@W^T.
// lanes 0-15: n=lane, k=k0..k0+7,k0+16..k0+23; lanes 16-31: n=lane-16, chunks +8.
__device__ __forceinline__ v16bf frag_b(const u16* W, int ldw, int n0, int k0){
  int lane = threadIdx.x & 31;
  const u16* p = W + (size_t)(n0 + (lane & 15)) * ldw + k0 + ((lane >> 4) << 3);
  FragU f;
  f.u8[0] = *(const ushort8*)p;
  f.u8[1] = *(const ushort8*)(p + 16);
  return f.bf;
}
__device__ __forceinline__ v8f wmma_bf(v16bf a, v16bf b, v8f c){
  return __builtin_amdgcn_wmma_f32_16x16x32_bf16(false, a, false, b, (short)0, c, false, false);
}

// ---------------------------------------------------------------------------
__global__ void k_cvt(const float* __restrict__ s, u16* __restrict__ d, int n){
  int i = blockIdx.x * 256 + threadIdx.x;
  if (i < n) d[i] = f2bf(s[i]);
}

// dst[(t*Bn+b)*128 + e] = bf16(emb[tok[b*S+t]*128+e])   (time-major rows)
__global__ void k_embed(const int* __restrict__ tok, const float* __restrict__ emb,
                        u16* __restrict__ dst, int Bn, int S, int n){
  int i = blockIdx.x * 256 + threadIdx.x;
  if (i >= n) return;
  int e = i & 127, r = i >> 7, t = r / Bn, b = r % Bn;
  dst[i] = f2bf(emb[(size_t)tok[b * S + t] * 128 + e]);
}

// Generic C(MxN) = A(MxK) @ W(NxK)^T, bf16 in / f32 out. 8 waves/block, 1 tile/wave.
__global__ void __launch_bounds__(256) k_gemm_xw(const u16* __restrict__ A,
                                                 const u16* __restrict__ W,
                                                 float* __restrict__ C,
                                                 int M, int N, int K){
  int wave = (int)(blockIdx.x * 8) + (threadIdx.x >> 5);
  int ntiles = N >> 4;
  if (wave >= (M >> 4) * ntiles) return;
  int m0 = (wave / ntiles) << 4, n0 = (wave % ntiles) << 4;
  v8f acc = {};
  for (int k = 0; k < K; k += 32)
    acc = wmma_bf(frag_a(A, K, m0, k), frag_b(W, K, n0, k), acc);
  int lane = threadIdx.x & 31;
  int n = n0 + (lane & 15), mb = m0 + ((lane >> 4) << 3);
#pragma unroll
  for (int r = 0; r < 8; ++r) C[(size_t)(mb + r) * N + n] = acc[r];
}

// Bidirectional encoder LSTM: block 0 = fwd, block 1 = bwd. 16 waves; wave w owns
// columns [w*16, w*16+16) of all four gate blocks -> cell update from registers.
__global__ void __launch_bounds__(512) k_enc_lstm(
    const float* __restrict__ Xf, const float* __restrict__ Xb,
    const u16* __restrict__ Whhf, const u16* __restrict__ Whhb,
    const float* __restrict__ bfv, const float* __restrict__ bbv,
    float* __restrict__ enc_hidden, u16* __restrict__ enc_hidden_bf,
    float* __restrict__ hN, float* __restrict__ cN){
  __shared__ float sC[16][256];
  __shared__ float sHf[16][256];
  __shared__ u16   sH[16][256];
  int dir = blockIdx.x;
  const float* X    = dir ? Xb : Xf;
  const u16*   Whh  = dir ? Whhb : Whhf;
  const float* bias = dir ? bbv : bfv;
  int tid = threadIdx.x, lane = tid & 31, w = tid >> 5;
  int col = w * 16 + (lane & 15);
  int mb = (lane >> 4) << 3;
  for (int i = tid; i < 4096; i += 512){
    int m = i >> 8, j = i & 255;
    sC[m][j] = 0.f; sHf[m][j] = 0.f; sH[m][j] = 0;
  }
  __syncthreads();
  for (int s = 0; s < 400; ++s){
    int t = dir ? (399 - s) : s;
    v8f acc[4];
#pragma unroll
    for (int g = 0; g < 4; ++g){
      int n0 = g * 256 + w * 16;
#pragma unroll
      for (int r = 0; r < 8; ++r)
        acc[g][r] = X[(size_t)(t * 16 + mb + r) * 1024 + n0 + (lane & 15)];
    }
    for (int k = 0; k < 256; k += 32){
      v16bf a = frag_a(&sH[0][0], 256, 0, k);
#pragma unroll
      for (int g = 0; g < 4; ++g)
        acc[g] = wmma_bf(a, frag_b(Whh, 256, g * 256 + w * 16, k), acc[g]);
    }
    __syncthreads();
#pragma unroll
    for (int r = 0; r < 8; ++r){
      int m = mb + r;
      float iv = sigm (acc[0][r] + bias[col]);
      float fv = sigm (acc[1][r] + bias[256 + col]);
      float gv = tanhf(acc[2][r] + bias[512 + col]);
      float ov = sigm (acc[3][r] + bias[768 + col]);
      float c = fv * sC[m][col] + iv * gv;
      float h = ov * tanhf(c);
      sC[m][col] = c; sHf[m][col] = h;
      u16 hb = f2bf(h);
      sH[m][col] = hb;
      size_t ei = ((size_t)(m * 400 + t)) * 512 + dir * 256 + col;
      enc_hidden[ei] = h;
      enc_hidden_bf[ei] = hb;
    }
    __syncthreads();
  }
  for (int i = tid; i < 4096; i += 512){
    int m = i >> 8, j = i & 255;
    hN[dir * 4096 + i] = sHf[m][j];
    cN[dir * 4096 + i] = sC[m][j];
  }
}

__global__ void __launch_bounds__(256) k_reduce(
    const float* __restrict__ hN, const float* __restrict__ cN,
    const float* __restrict__ rhW, const float* __restrict__ rhb,
    const float* __restrict__ rcW, const float* __restrict__ rcb,
    float* __restrict__ h_state, u16* __restrict__ h_bf, float* __restrict__ c_state){
  int j = threadIdx.x;
  for (int b = 0; b < 16; ++b){
    float hs = rhb[j], cs = rcb[j];
    for (int k = 0; k < 512; ++k){
      float hv = (k < 256) ? hN[b * 256 + k] : hN[4096 + b * 256 + (k - 256)];
      float cv = (k < 256) ? cN[b * 256 + k] : cN[4096 + b * 256 + (k - 256)];
      hs += rhW[j * 512 + k] * hv;
      cs += rcW[j * 512 + k] * cv;
    }
    hs = fmaxf(hs, 0.f); cs = fmaxf(cs, 0.f);
    h_state[b * 256 + j] = hs; h_bf[b * 256 + j] = f2bf(hs); c_state[b * 256 + j] = cs;
  }
}

// One decoder LSTM cell + dec_feat = h @ attn_Wdec^T + b.
__global__ void __launch_bounds__(512) k_dec_cell(
    const float* __restrict__ Xd, const u16* __restrict__ dWhh_bf,
    const float* __restrict__ dbv, const u16* __restrict__ Wdec_bf,
    const float* __restrict__ bdec, float* __restrict__ h_state,
    u16* __restrict__ h_bf, float* __restrict__ c_state,
    float* __restrict__ dec_feat, int t){
  __shared__ u16 sH[16][256];
  int tid = threadIdx.x, lane = tid & 31, w = tid >> 5;
  int col = w * 16 + (lane & 15), mb = (lane >> 4) << 3;
  v8f acc[4];
#pragma unroll
  for (int g = 0; g < 4; ++g){
    int n0 = g * 256 + w * 16;
#pragma unroll
    for (int r = 0; r < 8; ++r)
      acc[g][r] = Xd[(size_t)(t * 16 + mb + r) * 1024 + n0 + (lane & 15)];
  }
  for (int k = 0; k < 256; k += 32){
    v16bf a = frag_a(h_bf, 256, 0, k);
#pragma unroll
    for (int g = 0; g < 4; ++g)
      acc[g] = wmma_bf(a, frag_b(dWhh_bf, 256, g * 256 + w * 16, k), acc[g]);
  }
  __syncthreads();                       // everyone done reading prev h_bf
#pragma unroll
  for (int r = 0; r < 8; ++r){
    int m = mb + r;
    float iv = sigm (acc[0][r] + dbv[col]);
    float fv = sigm (acc[1][r] + dbv[256 + col]);
    float gv = tanhf(acc[2][r] + dbv[512 + col]);
    float ov = sigm (acc[3][r] + dbv[768 + col]);
    float c = fv * c_state[m * 256 + col] + iv * gv;
    float h = ov * tanhf(c);
    c_state[m * 256 + col] = c;
    h_state[m * 256 + col] = h;
    u16 hb = f2bf(h);
    h_bf[m * 256 + col] = hb;
    sH[m][col] = hb;
  }
  __syncthreads();
  v8f f0 = {}, f1 = {};
  for (int k = 0; k < 256; k += 32){
    v16bf a = frag_a(&sH[0][0], 256, 0, k);
    f0 = wmma_bf(a, frag_b(Wdec_bf, 256, w * 16, k), f0);
    f1 = wmma_bf(a, frag_b(Wdec_bf, 256, 256 + w * 16, k), f1);
  }
#pragma unroll
  for (int r = 0; r < 8; ++r){
    int m = mb + r;
    dec_feat[m * 512 + col]       = f0[r] + bdec[col];
    dec_feat[m * 512 + 256 + col] = f1[r] + bdec[256 + col];
  }
}

// Attention + softmax + context + p_gen, one block per batch element.
__global__ void __launch_bounds__(256) k_attn(
    const float* __restrict__ enc_feature, const float* __restrict__ enc_hidden,
    const float* __restrict__ dec_feat, const unsigned char* __restrict__ pad_mask,
    const float* __restrict__ attn_v, const float* __restrict__ h_state,
    const float* __restrict__ emb, const int* __restrict__ dec_input,
    const float* __restrict__ wh, const float* __restrict__ ws_v,
    const float* __restrict__ wx, const float* __restrict__ wxb,
    float* __restrict__ attn_out, float* __restrict__ ctx,
    u16* __restrict__ ctx_bf, float* __restrict__ p_gen, int t){
  __shared__ float sdf[512];
  __shared__ float sS[400];
  __shared__ float red[256];
  int b = blockIdx.x, tid = threadIdx.x, lane = tid & 31, w = tid >> 5;
  sdf[tid] = dec_feat[b * 512 + tid];
  sdf[tid + 256] = dec_feat[b * 512 + tid + 256];
  __syncthreads();
  for (int l = w; l < 400; l += 8){
    const float* ef = &enc_feature[((size_t)b * 400 + l) * 512 + lane * 16];
    float p = 0.f;
#pragma unroll
    for (int i = 0; i < 16; ++i) p += tanhf(ef[i] + sdf[lane * 16 + i]) * attn_v[lane * 16 + i];
    for (int off = 16; off; off >>= 1) p += __shfl_xor(p, off, 32);
    if (lane == 0) sS[l] = pad_mask[b * 400 + l] ? -INFINITY : p;
  }
  __syncthreads();
  float lm = -INFINITY;
  for (int l = tid; l < 400; l += 256) lm = fmaxf(lm, sS[l]);
  red[tid] = lm; __syncthreads();
  for (int s = 128; s; s >>= 1){ if (tid < s) red[tid] = fmaxf(red[tid], red[tid + s]); __syncthreads(); }
  float mx = red[0]; __syncthreads();
  float lsum = 0.f;
  for (int l = tid; l < 400; l += 256){ float e = expf(sS[l] - mx); sS[l] = e; lsum += e; }
  red[tid] = lsum; __syncthreads();
  for (int s = 128; s; s >>= 1){ if (tid < s) red[tid] += red[tid + s]; __syncthreads(); }
  float inv = 1.f / red[0]; __syncthreads();
  for (int l = tid; l < 400; l += 256){ float a = sS[l] * inv; sS[l] = a; attn_out[b * 400 + l] = a; }
  __syncthreads();
  float a0 = 0.f, a1 = 0.f;
  for (int l = 0; l < 400; ++l){
    float a = sS[l];
    const float* eh = &enc_hidden[((size_t)b * 400 + l) * 512];
    a0 += a * eh[tid]; a1 += a * eh[tid + 256];
  }
  ctx[b * 512 + tid] = a0;            ctx[b * 512 + tid + 256] = a1;
  ctx_bf[b * 512 + tid] = f2bf(a0);   ctx_bf[b * 512 + tid + 256] = f2bf(a1);
  float pp = a0 * wh[tid] + a1 * wh[tid + 256] + h_state[b * 256 + tid] * ws_v[tid];
  if (tid < 128){
    int tok = dec_input[b * 100 + t];
    pp += emb[(size_t)tok * 128 + tid] * wx[tid];
  }
  red[tid] = pp; __syncthreads();
  for (int s = 128; s; s >>= 1){ if (tid < s) red[tid] += red[tid + s]; __syncthreads(); }
  if (tid == 0) p_gen[b] = sigm(red[0] + wxb[0]);
}

// out(16x256) = [h | ctx](16x768) @ l1W^T + l1b  (bf16 out for vocab GEMM).
__global__ void __launch_bounds__(512) k_l1(const u16* __restrict__ h_bf,
                                            const u16* __restrict__ ctx_bf,
                                            const u16* __restrict__ l1W_bf,
                                            const float* __restrict__ l1b,
                                            u16* __restrict__ out_bf){
  __shared__ u16 sA[16][768];
  int tid = threadIdx.x, lane = tid & 31, w = tid >> 5;
  for (int i = tid; i < 4096; i += 512) sA[i >> 8][i & 255] = h_bf[i];
  for (int i = tid; i < 8192; i += 512) sA[i >> 9][256 + (i & 511)] = ctx_bf[i];
  __syncthreads();
  v8f acc = {};
  for (int k = 0; k < 768; k += 32)
    acc = wmma_bf(frag_a(&sA[0][0], 768, 0, k), frag_b(l1W_bf, 768, w * 16, k), acc);
  int col = w * 16 + (lane & 15), mb = (lane >> 4) << 3;
#pragma unroll
  for (int r = 0; r < 8; ++r) out_bf[(mb + r) * 256 + col] = f2bf(acc[r] + l1b[col]);
}

// logits(16x50000) = out @ l2W^T + l2b.  3125 N-tiles, 8 waves/block.
__global__ void __launch_bounds__(256) k_logits(const u16* __restrict__ out_bf,
                                                const u16* __restrict__ l2W_bf,
                                                const float* __restrict__ l2b,
                                                float* __restrict__ logits){
  int wave = (int)(blockIdx.x * 8) + (threadIdx.x >> 5);
  if (wave >= 3125) return;
  int lane = threadIdx.x & 31;
  int n0 = wave * 16;
  v8f acc = {};
  for (int k = 0; k < 256; k += 32)
    acc = wmma_bf(frag_a(out_bf, 256, 0, k), frag_b(l2W_bf, 256, n0, k), acc);
  int col = n0 + (lane & 15), mb = (lane >> 4) << 3;
#pragma unroll
  for (int r = 0; r < 8; ++r) logits[(size_t)(mb + r) * 50000 + col] = acc[r] + l2b[col];
}

__global__ void __launch_bounds__(256) k_rowstats(const float* __restrict__ logits,
                                                  float* __restrict__ rowmax,
                                                  float* __restrict__ rowsum){
  __shared__ float red[256];
  int b = blockIdx.x, tid = threadIdx.x;
  const float* lr = logits + (size_t)b * 50000;
  float lm = -INFINITY;
  for (int v = tid; v < 50000; v += 256) lm = fmaxf(lm, lr[v]);
  red[tid] = lm; __syncthreads();
  for (int s = 128; s; s >>= 1){ if (tid < s) red[tid] = fmaxf(red[tid], red[tid + s]); __syncthreads(); }
  float mx = red[0]; __syncthreads();
  float ls = 0.f;
  for (int v = tid; v < 50000; v += 256) ls += expf(lr[v] - mx);
  red[tid] = ls; __syncthreads();
  for (int s = 128; s; s >>= 1){ if (tid < s) red[tid] += red[tid + s]; __syncthreads(); }
  if (tid == 0){ rowmax[b] = mx; rowsum[b] = red[0]; }
}

// out[b, v, t] = p_gen*softmax(logits)[v] (v<V), 0 in OOV region.
__global__ void k_final(const float* __restrict__ logits, const float* __restrict__ rowmax,
                        const float* __restrict__ rowsum, const float* __restrict__ p_gen,
                        float* __restrict__ out, int t){
  int i = blockIdx.x * 256 + threadIdx.x;
  if (i >= 16 * 50100) return;
  int b = i / 50100, v = i % 50100;
  float val = 0.f;
  if (v < 50000)
    val = p_gen[b] * expf(logits[(size_t)b * 50000 + v] - rowmax[b]) / rowsum[b];
  out[((size_t)b * 50100 + v) * 100 + t] = val;
}

__global__ void k_scatter(const int* __restrict__ ext, const float* __restrict__ attn,
                          const float* __restrict__ p_gen, float* __restrict__ out, int t){
  int i = blockIdx.x * 256 + threadIdx.x;
  if (i >= 6400) return;
  int b = i / 400;
  int idx = ext[i];
  atomicAdd(&out[((size_t)b * 50100 + idx) * 100 + t], (1.f - p_gen[b]) * attn[i]);
}

// ---------------------------------------------------------------------------
extern "C" void kernel_launch(void* const* d_in, const int* in_sizes, int n_in,
                              void* d_out, int out_size, void* d_ws, size_t ws_size,
                              hipStream_t stream){
  (void)in_sizes; (void)n_in; (void)out_size; (void)ws_size;
  const int*   enc_input     = (const int*)d_in[0];
  const int*   enc_input_ext = (const int*)d_in[1];
  const unsigned char* pad_mask = (const unsigned char*)d_in[2];
  const int*   dec_input     = (const int*)d_in[3];
  const float* emb   = (const float*)d_in[5];
  const float* Wih_f = (const float*)d_in[6];
  const float* Whh_f = (const float*)d_in[7];
  const float* b_f   = (const float*)d_in[8];
  const float* Wih_b = (const float*)d_in[9];
  const float* Whh_b = (const float*)d_in[10];
  const float* b_b   = (const float*)d_in[11];
  const float* red_hW = (const float*)d_in[12];
  const float* red_hb = (const float*)d_in[13];
  const float* red_cW = (const float*)d_in[14];
  const float* red_cb = (const float*)d_in[15];
  const float* attn_v    = (const float*)d_in[16];
  const float* attn_Wenc = (const float*)d_in[17];
  const float* attn_Wdec = (const float*)d_in[18];
  const float* attn_bdec = (const float*)d_in[19];
  const float* dWih = (const float*)d_in[20];
  const float* dWhh = (const float*)d_in[21];
  const float* dbv  = (const float*)d_in[22];
  const float* l1W  = (const float*)d_in[23];
  const float* l1b  = (const float*)d_in[24];
  const float* l2W  = (const float*)d_in[25];
  const float* l2b  = (const float*)d_in[26];
  const float* wh   = (const float*)d_in[27];
  const float* ws_v = (const float*)d_in[28];
  const float* wx   = (const float*)d_in[29];
  const float* wxb  = (const float*)d_in[30];
  float* out = (float*)d_out;

  size_t off = 0;
  auto alloc = [&](size_t bytes) -> void* {
    void* p = (char*)d_ws + off;
    off = (off + bytes + 255) & ~(size_t)255;
    return p;
  };
  u16* bWihf = (u16*)alloc(131072u * 2);
  u16* bWhhf = (u16*)alloc(262144u * 2);
  u16* bWihb = (u16*)alloc(131072u * 2);
  u16* bWhhb = (u16*)alloc(262144u * 2);
  u16* bdWih = (u16*)alloc(131072u * 2);
  u16* bdWhh = (u16*)alloc(262144u * 2);
  u16* bWenc = (u16*)alloc(262144u * 2);
  u16* bWdec = (u16*)alloc(131072u * 2);
  u16* bl1W  = (u16*)alloc(196608u * 2);
  u16* bl2W  = (u16*)alloc((size_t)12800000u * 2);
  u16* xs_bf = (u16*)alloc(819200u * 2);
  u16* xd_bf = (u16*)alloc(204800u * 2);
  float* Xf = (float*)alloc((size_t)6553600u * 4);
  float* Xb = (float*)alloc((size_t)6553600u * 4);
  float* Xd = (float*)alloc((size_t)1638400u * 4);
  float* enc_hidden    = (float*)alloc((size_t)3276800u * 4);
  u16*   enc_hidden_bf = (u16*)  alloc((size_t)3276800u * 2);
  float* enc_feature   = (float*)alloc((size_t)3276800u * 4);
  float* hN = (float*)alloc(8192u * 4);
  float* cN = (float*)alloc(8192u * 4);
  float* h_state = (float*)alloc(4096u * 4);
  u16*   h_bf    = (u16*)alloc(4096u * 2);
  float* c_state = (float*)alloc(4096u * 4);
  float* dec_feat = (float*)alloc(8192u * 4);
  float* attn_buf = (float*)alloc(6400u * 4);
  float* ctx      = (float*)alloc(8192u * 4);
  u16*   ctx_bf   = (u16*)alloc(8192u * 2);
  float* p_gen    = (float*)alloc(64u);
  u16*   out_bf   = (u16*)alloc(4096u * 2);
  float* logits   = (float*)alloc((size_t)800000u * 4);
  float* rowmax   = (float*)alloc(64u);
  float* rowsum   = (float*)alloc(64u);

  auto cvt = [&](const float* s, u16* d, int n){
    k_cvt<<<(n + 255) / 256, 256, 0, stream>>>(s, d, n);
  };
  cvt(Wih_f, bWihf, 131072);  cvt(Whh_f, bWhhf, 262144);
  cvt(Wih_b, bWihb, 131072);  cvt(Whh_b, bWhhb, 262144);
  cvt(dWih,  bdWih, 131072);  cvt(dWhh,  bdWhh, 262144);
  cvt(attn_Wenc, bWenc, 262144);
  cvt(attn_Wdec, bWdec, 131072);
  cvt(l1W, bl1W, 196608);
  cvt(l2W, bl2W, 12800000);

  k_embed<<<(819200 + 255) / 256, 256, 0, stream>>>(enc_input, emb, xs_bf, 16, 400, 819200);
  k_embed<<<(204800 + 255) / 256, 256, 0, stream>>>(dec_input, emb, xd_bf, 16, 100, 204800);

  auto gemm = [&](const u16* A, const u16* W, float* C, int M, int N, int K){
    int waves = (M / 16) * (N / 16);
    k_gemm_xw<<<(waves + 7) / 8, 256, 0, stream>>>(A, W, C, M, N, K);
  };
  gemm(xs_bf, bWihf, Xf, 6400, 1024, 128);
  gemm(xs_bf, bWihb, Xb, 6400, 1024, 128);
  gemm(xd_bf, bdWih, Xd, 1600, 1024, 128);

  k_enc_lstm<<<2, 512, 0, stream>>>(Xf, Xb, bWhhf, bWhhb, b_f, b_b,
                                    enc_hidden, enc_hidden_bf, hN, cN);
  gemm(enc_hidden_bf, bWenc, enc_feature, 6400, 512, 512);
  k_reduce<<<1, 256, 0, stream>>>(hN, cN, red_hW, red_hb, red_cW, red_cb,
                                  h_state, h_bf, c_state);

  for (int t = 0; t < 100; ++t){
    k_dec_cell<<<1, 512, 0, stream>>>(Xd, bdWhh, dbv, bWdec, attn_bdec,
                                      h_state, h_bf, c_state, dec_feat, t);
    k_attn<<<16, 256, 0, stream>>>(enc_feature, enc_hidden, dec_feat, pad_mask,
                                   attn_v, h_state, emb, dec_input, wh, ws_v,
                                   wx, wxb, attn_buf, ctx, ctx_bf, p_gen, t);
    k_l1<<<1, 512, 0, stream>>>(h_bf, ctx_bf, bl1W, l1b, out_bf);
    k_logits<<<391, 256, 0, stream>>>(out_bf, bl2W, l2b, logits);
    k_rowstats<<<16, 256, 0, stream>>>(logits, rowmax, rowsum);
    k_final<<<(16 * 50100 + 255) / 256, 256, 0, stream>>>(logits, rowmax, rowsum,
                                                          p_gen, out, t);
    k_scatter<<<25, 256, 0, stream>>>(enc_input_ext, attn_buf, p_gen, out, t);
  }
}
